// NegativeDictionary_58935541236465
// MI455X (gfx1250) — matrix operand order repaired and checked by reading, not compiled, not run on hardware
//
#include <hip/hip_runtime.h>
#include <stdint.h>

typedef __attribute__((ext_vector_type(16))) __bf16 v16bf;
typedef __attribute__((ext_vector_type(8)))  float  v8f;

#define CC 512          // channels (K)
#define SS 4096         // H*W
#define NN 512          // dict rows
#define THRESH 0.625f

__device__ __forceinline__ unsigned short f2bf(float f){
    unsigned u = __float_as_uint(f);
    u += 0x7FFFu + ((u >> 16) & 1u);           // round-to-nearest-even
    return (unsigned short)(u >> 16);
}

// ---------------- kernel 1: reciprocal row norms of the dictionary ----------------
__global__ void k_rnorm(const float* __restrict__ dict, float* __restrict__ rnorm){
    __shared__ float red[256];
    int n = blockIdx.x, t = threadIdx.x;
    float s = 0.f;
    for (int c = t; c < CC; c += 256){ float v = dict[n*CC + c]; s += v*v; }
    red[t] = s; __syncthreads();
    for (int off = 128; off > 0; off >>= 1){
        if (t < off) red[t] += red[t+off];
        __syncthreads();
    }
    if (t == 0) rnorm[n] = rsqrtf(red[0]);
}

// ---------------- kernel 2: pack normalized dict (bf16) into WMMA B layout -------
// packed[t][q][lane][i] : n = t*16 + lane%16 , k = q*32 + i + 16*(lane/16)
__global__ void k_pack(const float* __restrict__ dict, const float* __restrict__ rnorm,
                       unsigned short* __restrict__ packed){
    int o = blockIdx.x*256 + threadIdx.x;      // 0 .. 262143
    int i    = o & 15;
    int lane = (o >> 4) & 31;
    int q    = (o >> 9) & 15;
    int t    =  o >> 13;
    int n = t*16 + (lane & 15);
    int k = q*32 + i + ((lane >> 4) << 4);
    packed[o] = f2bf(dict[n*CC + k] * rnorm[n]);
}

// ---------------- kernel 3: fused normalize + WMMA GEMM + argmin + blend ---------
__global__ void __launch_bounds__(128) k_main(const float* __restrict__ x,
                                              const float* __restrict__ dict,
                                              const unsigned short* __restrict__ packedB,
                                              float* __restrict__ out){
    __shared__ unsigned int bLds[2][4096];     // 2 x 16 KiB: double-buffered B tiles
    __shared__ float dminLds[64];
    __shared__ int   idxLds[64];

    const int tid  = threadIdx.x;
    const int wid  = tid >> 5;
    const int lane = tid & 31;
    const int m    = lane & 15;                // position within M-tile
    const int half = lane >> 4;

    const int tile = blockIdx.x*4 + wid;       // M-tile id
    const int p0 = tile << 4;                  // first flat spatial position
    const int b  = p0 >> 12;                   // batch (4096 positions each)
    const int s0 = p0 & 4095;
    const int xBase = b*CC*SS + s0;            // + c*SS + m

    // phase 1: per-position sum of squares (lanes split channel halves, coalesced in m)
    float sum = 0.f;
    {
        int c0 = half*256;
        for (int c = c0; c < c0+256; ++c){
            float v = x[xBase + c*SS + m];
            sum += v*v;
        }
    }
    sum += __shfl_xor(sum, 16, 32);
    float rn = rsqrtf(sum);

    // phase 2: build A fragment entirely in VGPRs, contiguous k order (same as packed B)
    // K(i,half) = i + 16*half
    v16bf aFrag[16];
    #pragma unroll
    for (int q = 0; q < 16; ++q){
        union { unsigned u[8]; v16bf v; } tmp;
        int kb = q*32 + half*16;
        #pragma unroll
        for (int ip = 0; ip < 8; ++ip){
            int i = ip*2;
            int k = kb + i;
            float v0 = x[xBase + k*SS + m] * rn;
            float v1 = x[xBase + (k+1)*SS + m] * rn;
            tmp.u[ip] = (unsigned)f2bf(v0) | ((unsigned)f2bf(v1) << 16);
        }
        aFrag[q] = tmp.v;
    }

    float minv[8]; int mini[8];
    #pragma unroll
    for (int r = 0; r < 8; ++r){ minv[r] = 3.0e38f; mini[r] = 0; }

    const uint4* pB = (const uint4*)packedB;

    // async-copy one 16 KiB B tile into LDS buffer `buf` (8 x b128 per thread)
    auto issueTile = [&](int tt, int buf){
        const uint4* src = pB + tt*1024;
        uint4* dst = (uint4*)bLds[buf];
        #pragma unroll
        for (int j = 0; j < 8; ++j){
            int idx = tid + j*128;
            unsigned ldsOff = (unsigned)(uintptr_t)(dst + idx);
            unsigned long long g = (unsigned long long)(uintptr_t)(src + idx);
            asm volatile("global_load_async_to_lds_b128 %0, %1, off"
                         :: "v"(ldsOff), "v"(g) : "memory");
        }
    };

    issueTile(0, 0);                            // prologue: tile 0 in flight

    for (int t = 0; t < 32; ++t){
        __syncthreads();                        // everyone done reading buf[(t+1)&1]
        if (t + 1 < 32){
            issueTile(t+1, (t+1)&1);            // overlap next tile with compute
            // 8 newer asyncs outstanding => tile t fully landed when cnt<=8
            asm volatile("s_wait_asynccnt 0x8" ::: "memory");
        } else {
            asm volatile("s_wait_asynccnt 0x0" ::: "memory");
        }
        __syncthreads();                        // tile t visible to all waves

        const unsigned int* bCur = bLds[t & 1];
        v8f acc = {0.f,0.f,0.f,0.f,0.f,0.f,0.f,0.f};
        #pragma unroll
        for (int q = 0; q < 16; ++q){
            v16bf bv = *(const v16bf*)&bCur[q*256 + lane*8];
            acc = __builtin_amdgcn_wmma_f32_16x16x32_bf16(false, aFrag[q], false, bv,
                                                          (short)0, acc, false, false);
        }
        // C layout: VGPR r -> M = r + 8*half ; N = t*16 + m
        int n = t*16 + m;
        #pragma unroll
        for (int r = 0; r < 8; ++r){
            float dist = 0.5f - 0.5f*acc[r];
            bool lt = dist < minv[r];
            minv[r] = lt ? dist : minv[r];
            mini[r] = lt ? n    : mini[r];
        }
    }

    // reduce argmin across the 16 lanes of each half (masks stay within half)
    #pragma unroll
    for (int mask = 1; mask < 16; mask <<= 1){
        #pragma unroll
        for (int r = 0; r < 8; ++r){
            float ov = __shfl_xor(minv[r], mask, 32);
            int   oi = __shfl_xor(mini[r], mask, 32);
            bool lt = ov < minv[r];
            minv[r] = lt ? ov : minv[r];
            mini[r] = lt ? oi : mini[r];
        }
    }
    __syncthreads();
    if (m == 0){
        #pragma unroll
        for (int r = 0; r < 8; ++r){
            dminLds[wid*16 + half*8 + r] = minv[r];
            idxLds [wid*16 + half*8 + r] = mini[r];
        }
    }
    __syncthreads();

    // phase 5: blended output (replace feature by nearest dict row if dist < thresh)
    float dmin = dminLds[wid*16 + m];
    int   idx  = idxLds [wid*16 + m];
    bool  rep  = dmin < THRESH;
    const float* drow = dict + idx*CC;
    for (int it = 0; it < 256; ++it){
        int c = it*2 + half;
        int a = xBase + c*SS + m;
        float v = rep ? drow[c] : x[a];
        out[a] = v;
    }
}

extern "C" void kernel_launch(void* const* d_in, const int* in_sizes, int n_in,
                              void* d_out, int out_size, void* d_ws, size_t ws_size,
                              hipStream_t stream){
    const float* x    = (const float*)d_in[0];    // [16,512,64,64]
    const float* dict = (const float*)d_in[1];    // [512,512]
    float* out = (float*)d_out;

    float* rnorm = (float*)d_ws;                                    // 2 KiB
    unsigned short* packed = (unsigned short*)((char*)d_ws + 4096); // 512 KiB

    k_rnorm<<<512, 256, 0, stream>>>(dict, rnorm);
    k_pack <<<1024,256, 0, stream>>>(dict, rnorm, packed);
    k_main <<<1024,128, 0, stream>>>(x, dict, packed, out);
}
